// RNN_8821862826329
// MI455X (gfx1250) — matrix operand order repaired and checked by reading, not compiled, not run on hardware
//
#include <hip/hip_runtime.h>

// PredRNN++ (CausalLSTM + GHU) forward for MI455X (gfx1250, wave32).
// All convs -> implicit GEMM with V_WMMA_F32_16X16X32_F16 (f16 in, f32 acc).
// Activations kept channel-major [Ch, N] with N = B*H*W = 2048.
// GEMM: each wave computes a 16(M) x 64(N) strip -> 4 accumulators share one
// A (weight) fragment per K step: 10 b128 loads per 4 WMMAs.

typedef __attribute__((ext_vector_type(16))) _Float16 v16h;
typedef __attribute__((ext_vector_type(8)))  _Float16 v8h;
typedef __attribute__((ext_vector_type(8)))  float    v8f;

namespace {
constexpr int BB = 8, TT = 20, CC = 16, NHID = 64;
constexpr int HWs = 256;                   // 16*16
constexpr int NN = BB * HWs;               // 2048 pixel columns
constexpr int S  = NHID * NN;              // one NH-channel plane (131072 floats)
constexpr int INPUT_LEN = 10;
constexpr int TSTEPS = TT - 1;             // 19
constexpr int GEN_ELEMS = BB * TSTEPS * CC * HWs; // 622592
}

__device__ __forceinline__ float sigm(float x) { return 1.0f / (1.0f + __expf(-x)); }

// ---------------- WMMA GEMM: D[M,N] = A[M,Kp] * Bcol[N,Kp]^T ----------------
// Fragment layouts (A/B: contiguous k order used by this code; C/D per CDNA5 ISA 7.12.2):
//  A (16x32 f16): lanes 0-15 hold M=lane, K = k0+0..15;
//                 lanes 16-31 hold K = k0+16..31.
//  B (32x16 f16): lanes 0-15 hold N=lane, K = k0+0..15; lanes 16-31 K=k0+16..31.
//  C/D f32: VGPR r, lane l -> M = r + 8*(l>=16), N = l%16.
__launch_bounds__(256)
__global__ void gemm_wmma_k(const _Float16* __restrict__ A,
                            const _Float16* __restrict__ Bm,
                            float* __restrict__ D,
                            int M, int Kp)
{
  const int wave  = blockIdx.x * (blockDim.x >> 5) + (threadIdx.x >> 5);
  const int lane  = threadIdx.x & 31;
  const int ng    = wave & 31;        // 32 groups of 4 N-tiles (64 columns)
  const int mtile = wave >> 5;
  if (mtile * 16 >= M) return;        // wave-uniform: EXEC stays all-ones
  const int lh = lane >> 4;
  const int lm = lane & 15;
  const _Float16* aP = A  + (size_t)(mtile * 16 + lm) * Kp + lh * 16;
  const _Float16* bP = Bm + (size_t)(ng * 64 + lm) * Kp + lh * 16;
  const size_t bS = (size_t)16 * Kp;  // one N-tile stride in col buffer
  v8f acc0 = {}, acc1 = {}, acc2 = {}, acc3 = {};
  for (int k0 = 0; k0 < Kp; k0 += 32) {
    // prefetch the (heavily reused) weight row 512 halves (16 K-blocks) ahead
    __builtin_prefetch((const void*)(aP + k0 + 512), 0, 3);
    v8h alo = *(const v8h*)(aP + k0);
    v8h ahi = *(const v8h*)(aP + k0 + 8);
    v16h a;
#pragma unroll
    for (int i = 0; i < 8; ++i) { a[i] = alo[i]; a[i + 8] = ahi[i]; }
    v16h b0 = *(const v16h*)(bP + k0);
    v16h b1 = *(const v16h*)(bP + bS + k0);
    v16h b2 = *(const v16h*)(bP + 2 * bS + k0);
    v16h b3 = *(const v16h*)(bP + 3 * bS + k0);
    acc0 = __builtin_amdgcn_wmma_f32_16x16x32_f16(false, a, false, b0,
                                                  (short)0, acc0, false, false);
    acc1 = __builtin_amdgcn_wmma_f32_16x16x32_f16(false, a, false, b1,
                                                  (short)0, acc1, false, false);
    acc2 = __builtin_amdgcn_wmma_f32_16x16x32_f16(false, a, false, b2,
                                                  (short)0, acc2, false, false);
    acc3 = __builtin_amdgcn_wmma_f32_16x16x32_f16(false, a, false, b3,
                                                  (short)0, acc3, false, false);
  }
  float* dP = D + (size_t)(mtile * 16 + lh * 8) * NN + ng * 64 + lm;
#pragma unroll
  for (int r = 0; r < 8; ++r) {
    dP[(size_t)r * NN]      = acc0[r];
    dP[(size_t)r * NN + 16] = acc1[r];
    dP[(size_t)r * NN + 32] = acc2[r];
    dP[(size_t)r * NN + 48] = acc3[r];
  }
}

// ---------------- im2col: X[Cin,2048] f32 -> col[2048, Kp] f16 ----------------
__global__ void im2col_k(const float* __restrict__ X, _Float16* __restrict__ col,
                         int Cin, int KHW, int Kp)
{
  int idx = blockIdx.x * blockDim.x + threadIdx.x;
  if (idx >= NN * Kp) return;
  int n = idx / Kp;
  int k = idx - n * Kp;
  _Float16 v = (_Float16)0.0f;
  if (k < Cin * KHW) {
    if (KHW == 1) {
      v = (_Float16)X[(size_t)k * NN + n];
    } else {
      int ci = k / 25, rem = k - ci * 25;
      int kh = rem / 5, kw = rem - kh * 5;
      int b = n >> 8, p = n & 255, y = p >> 4, x = p & 15;
      int yy = y + kh - 2, xx = x + kw - 2;
      if ((unsigned)yy < 16u && (unsigned)xx < 16u)
        v = (_Float16)X[(size_t)ci * NN + (b << 8) + (yy << 4) + xx];
    }
  }
  col[idx] = v;
}

// ---------------- weight pack: f32 [Cout, Cin*KHW] -> f16 [Cout, Kp] ----------
__global__ void packw_k(const float* __restrict__ w, _Float16* __restrict__ dst,
                        int cout, int kreal, int kp)
{
  int i = blockIdx.x * blockDim.x + threadIdx.x;
  if (i >= cout * kp) return;
  int co = i / kp, k = i - co * kp;
  dst[i] = (k < kreal) ? (_Float16)w[(size_t)co * kreal + k] : (_Float16)0.0f;
}

__global__ void zero_k(float* __restrict__ p, int n)
{
  int i = blockIdx.x * blockDim.x + threadIdx.x;
  if (i < n) p[i] = 0.0f;
}

// ------------- tensor layernorm over (Ch, HW) per sample, in place ------------
__global__ void ln_k(float* __restrict__ buf, int Ch)
{
  const int b = blockIdx.x;          // 0..7
  const int tid = threadIdx.x;       // 256 threads
  __shared__ float s1[256], s2[256];
  const int count = Ch * 256;
  float sum = 0.f, sq = 0.f;
  for (int i = tid; i < count; i += 256) {
    int row = i >> 8, cc = i & 255;
    float v = buf[(size_t)row * NN + (b << 8) + cc];
    sum += v; sq += v * v;
  }
  s1[tid] = sum; s2[tid] = sq;
  __syncthreads();
  for (int off = 128; off > 0; off >>= 1) {
    if (tid < off) { s1[tid] += s1[tid + off]; s2[tid] += s2[tid + off]; }
    __syncthreads();
  }
  float mean = s1[0] / count;
  float var  = s2[0] / count - mean * mean;
  float rstd = rsqrtf(var + 1e-5f);
  for (int i = tid; i < count; i += 256) {
    int row = i >> 8, cc = i & 255;
    size_t o = (size_t)row * NN + (b << 8) + cc;
    buf[o] = (buf[o] - mean) * rstd;
  }
}

// ----------------------------- gate fusions ----------------------------------
__global__ void stage1_k(const float* __restrict__ lnx, const float* __restrict__ lnh,
                         const float* __restrict__ lnc, float* __restrict__ c)
{
  int i = blockIdx.x * blockDim.x + threadIdx.x;
  if (i >= S) return;
  float ig = sigm (lnx[i]         + lnh[i]         + lnc[i]);
  float g  = tanhf(lnx[i +   S]   + lnh[i +   S]   + lnc[i +   S]);
  float f  = sigm (lnx[i + 2 * S] + lnh[i + 2 * S] + lnc[i + 2 * S] + 1.0f);
  c[i] = f * c[i] + ig * g;                 // c_new in place
}

__global__ void stage2_k(const float* __restrict__ lnc2, const float* __restrict__ lnx,
                         const float* __restrict__ lnm, float* __restrict__ m)
{
  int i = blockIdx.x * blockDim.x + threadIdx.x;
  if (i >= S) return;
  float ii = sigm (lnc2[i]         + lnx[i + 4 * S] + lnm[i]);
  float gg = tanhf(lnc2[i +   S]   + lnx[i + 5 * S]);
  float ff = sigm (lnc2[i + 2 * S] + lnx[i + 6 * S] + lnm[i + S] + 1.0f);
  m[i] = ff * tanhf(lnm[i + 2 * S]) + ii * gg;   // m_new in place
}

__global__ void stage3_k(const float* __restrict__ lnx, const float* __restrict__ lnh,
                         const float* __restrict__ lnc2, const float* __restrict__ om,
                         const float* __restrict__ cell, float* __restrict__ h)
{
  int i = blockIdx.x * blockDim.x + threadIdx.x;
  if (i >= S) return;
  float o = tanhf(lnx[i + 3 * S] + lnh[i + 3 * S] + lnc2[i + 3 * S] + om[i]);
  h[i] = o * tanhf(cell[i]);
}

__global__ void ghu_k(const float* __restrict__ lnz, const float* __restrict__ lnxg,
                      float* __restrict__ z)
{
  int i = blockIdx.x * blockDim.x + threadIdx.x;
  if (i >= S) return;
  float p = tanhf(lnxg[i]     + lnz[i]);
  float u = sigm (lnxg[i + S] + lnz[i + S]);
  z[i] = u * p + (1.0f - u) * z[i];
}

__global__ void cat_k(const float* __restrict__ c, const float* __restrict__ m,
                      float* __restrict__ cat)
{
  int i = blockIdx.x * blockDim.x + threadIdx.x;
  if (i >= 2 * S) return;
  cat[i] = (i < S) ? c[i] : m[i - S];
}

// -------------------- scheduled-sampling input blend --------------------------
__global__ void blend_k(const float* __restrict__ images, const float* __restrict__ mask,
                        const float* __restrict__ xgen, float* __restrict__ xin, int t)
{
  int i = blockIdx.x * blockDim.x + threadIdx.x;    // exactly CC*NN = 32768
  int cch = i >> 11;
  int n = i & (NN - 1);
  int b = n >> 8, p = n & 255;
  float img = images[(((size_t)b * TT + t) * CC + cch) * HWs + p];
  float v = img;
  if (t >= INPUT_LEN) {
    float m = mask[(((size_t)b * (TT - INPUT_LEN - 1) + (t - INPUT_LEN)) * CC + cch) * HWs + p];
    v = m * img + (1.0f - m) * xgen[i];
  }
  xin[i] = v;
}

// ------------- x_gen = conv + bias; emit gen*255 and MSE partial --------------
__global__ void output_k(const float* __restrict__ conv, const float* __restrict__ bias,
                         const float* __restrict__ images, float* __restrict__ xgen,
                         float* __restrict__ gen, float* __restrict__ loss, int t)
{
  int i = blockIdx.x * blockDim.x + threadIdx.x;    // exactly CC*NN = 32768
  int cch = i >> 11;
  int n = i & (NN - 1);
  int b = n >> 8, p = n & 255;
  float v = conv[i] + bias[cch];
  xgen[i] = v;
  gen[(((size_t)b * TSTEPS + t) * CC + cch) * HWs + p] = v * 255.0f;
  float d = v - images[(((size_t)b * TT + (t + 1)) * CC + cch) * HWs + p];
  __shared__ float sred[256];
  sred[threadIdx.x] = d * d;
  __syncthreads();
  for (int off = 128; off > 0; off >>= 1) {
    if ((int)threadIdx.x < off) sred[threadIdx.x] += sred[threadIdx.x + off];
    __syncthreads();
  }
  if (threadIdx.x == 0) atomicAdd(loss, sred[0] * (1.0f / (float)GEN_ELEMS));
}

// =============================== host side ===================================
static inline int ceil_div(int a, int b) { return (a + b - 1) / b; }
static inline int kpad(int k) { return (k + 31) & ~31; }

extern "C" void kernel_launch(void* const* d_in, const int* in_sizes, int n_in,
                              void* d_out, int out_size, void* d_ws, size_t ws_size,
                              hipStream_t stream)
{
  (void)in_sizes; (void)n_in; (void)out_size; (void)ws_size;

  // setup_inputs() dict insertion order:
  //   images, mask_true, then params: layers[0..3]{Wx,Wh,Wc,Wm,Wc2m,Wom,Wcell},
  //   ghu_z, ghu_x, Wout, bout.
  const float* images = (const float*)d_in[0];
  const float* mask   = (const float*)d_in[1];
  const float *Wx[4], *Wh[4], *Wc[4], *Wm[4], *Wc2m[4], *Wom[4], *Wcell[4];
  int pi = 2;
  for (int i = 0; i < 4; ++i) {
    Wx[i]   = (const float*)d_in[pi++]; Wh[i]  = (const float*)d_in[pi++];
    Wc[i]   = (const float*)d_in[pi++]; Wm[i]  = (const float*)d_in[pi++];
    Wc2m[i] = (const float*)d_in[pi++]; Wom[i] = (const float*)d_in[pi++];
    Wcell[i]= (const float*)d_in[pi++];
  }
  const float* GzW   = (const float*)d_in[pi++];
  const float* GxW   = (const float*)d_in[pi++];
  const float* WoutW = (const float*)d_in[pi++];
  const float* boutW = (const float*)d_in[pi++];

  float* gen  = (float*)d_out;
  float* loss = gen + GEN_ELEMS;

  // ---- workspace bump allocator (256B aligned) ----
  char* ws = (char*)d_ws;
  size_t off = 0;
  auto allocF = [&](size_t n)->float* {
    float* p = (float*)(ws + off); off += ((n * 4 + 255) & ~(size_t)255); return p; };
  auto allocH = [&](size_t n)->_Float16* {
    _Float16* p = (_Float16*)(ws + off); off += ((n * 2 + 255) & ~(size_t)255); return p; };

  // persistent state, channel-major [Ch, 2048]
  float* h4   = allocF(4 * (size_t)S);
  float* c4   = allocF(4 * (size_t)S);
  float* memb = allocF(S);
  float* zb   = allocF(S);
  float* xgen = allocF(CC * NN);
  // temporaries (conv outputs double as LN buffers, normalized in place)
  float* xin   = allocF(CC * NN);
  float* lnx   = allocF(7 * (size_t)S);
  float* lnh   = allocF(4 * (size_t)S);
  float* lnc   = allocF(3 * (size_t)S);
  float* lnm   = allocF(3 * (size_t)S);
  float* lnc2  = allocF(4 * (size_t)S);
  float* omb   = allocF(S);
  float* cellb = allocF(S);
  float* catb  = allocF(2 * (size_t)S);
  float* lnzb  = allocF(2 * (size_t)S);
  float* lnxg  = allocF(2 * (size_t)S);
  float* outc  = allocF(CC * NN);

  // ---- pack all weights to f16 [Cout, Kpad] (every call; deterministic) ----
  auto pack = [&](const float* src, int cout, int cin, int khw)->_Float16* {
    int kp = kpad(cin * khw);
    _Float16* dst = allocH((size_t)cout * kp);
    int total = cout * kp;
    packw_k<<<ceil_div(total, 256), 256, 0, stream>>>(src, dst, cout, cin * khw, kp);
    return dst;
  };
  _Float16 *pWx[4], *pWh[4], *pWc[4], *pWm[4], *pWc2m[4], *pWom[4], *pWcell[4];
  for (int i = 0; i < 4; ++i) {
    int cinx = (i == 0) ? CC : NHID;
    pWx[i]    = pack(Wx[i],   7 * NHID, cinx, 25);
    pWh[i]    = pack(Wh[i],   4 * NHID, NHID, 25);
    pWc[i]    = pack(Wc[i],   3 * NHID, NHID, 25);
    pWm[i]    = pack(Wm[i],   3 * NHID, NHID, 25);
    pWc2m[i]  = pack(Wc2m[i], 4 * NHID, NHID, 25);
    pWom[i]   = pack(Wom[i],  NHID,     NHID, 25);
    pWcell[i] = pack(Wcell[i],NHID, 2 * NHID, 1);
  }
  _Float16* pGz   = pack(GzW,   2 * NHID, NHID, 25);
  _Float16* pGx   = pack(GxW,   2 * NHID, NHID, 25);
  _Float16* pWout = pack(WoutW, CC,       NHID, 1);

  _Float16* col = allocH((size_t)NN * 1600);   // largest Kpad (64*25)

  // ---- helpers ----
  auto zero = [&](float* p, int n) {
    zero_k<<<ceil_div(n, 256), 256, 0, stream>>>(p, n); };
  auto conv = [&](const float* x, int cin, int khw, const _Float16* wp,
                  int cout, float* out) {
    int kp = kpad(cin * khw);
    im2col_k<<<ceil_div(NN * kp, 256), 256, 0, stream>>>(x, col, cin, khw, kp);
    // waves = (cout/16) M-tiles * 32 N-groups; 8 waves per 256-thread block
    gemm_wmma_k<<<(cout / 16) * 4, 256, 0, stream>>>(wp, col, out, cout, kp);
  };
  auto lnrun = [&](float* buf, int ch) { ln_k<<<BB, 256, 0, stream>>>(buf, ch); };

  // ---- init state + loss ----
  zero(h4, 4 * S); zero(c4, 4 * S); zero(memb, S); zero(zb, S);
  zero(xgen, CC * NN); zero(loss, 1);

  auto lstm = [&](int i, const float* xp, int cinx) {
    conv(xp,         cinx, 25, pWx[i],   7 * NHID, lnx);  lnrun(lnx,  7 * NHID);
    conv(h4 + (size_t)i * S, NHID, 25, pWh[i], 4 * NHID, lnh); lnrun(lnh, 4 * NHID);
    conv(c4 + (size_t)i * S, NHID, 25, pWc[i], 3 * NHID, lnc); lnrun(lnc, 3 * NHID);
    conv(memb,       NHID, 25, pWm[i],   3 * NHID, lnm);  lnrun(lnm,  3 * NHID);
    stage1_k<<<S / 256, 256, 0, stream>>>(lnx, lnh, lnc, c4 + (size_t)i * S);
    conv(c4 + (size_t)i * S, NHID, 25, pWc2m[i], 4 * NHID, lnc2); lnrun(lnc2, 4 * NHID);
    stage2_k<<<S / 256, 256, 0, stream>>>(lnc2, lnx, lnm, memb);
    conv(memb,       NHID, 25, pWom[i],  NHID,     omb);   // no LN on o_m
    cat_k<<<2 * S / 256, 256, 0, stream>>>(c4 + (size_t)i * S, memb, catb);
    conv(catb, 2 * NHID, 1, pWcell[i], NHID, cellb);
    stage3_k<<<S / 256, 256, 0, stream>>>(lnx, lnh, lnc2, omb, cellb,
                                          h4 + (size_t)i * S);
  };

  // ---- time loop ----
  for (int t = 0; t < TSTEPS; ++t) {
    blend_k<<<CC * NN / 256, 256, 0, stream>>>(images, mask, xgen, xin, t);
    lstm(0, xin, CC);
    conv(zb, NHID, 25, pGz, 2 * NHID, lnzb); lnrun(lnzb, 2 * NHID);
    conv(h4, NHID, 25, pGx, 2 * NHID, lnxg); lnrun(lnxg, 2 * NHID);
    ghu_k<<<S / 256, 256, 0, stream>>>(lnzb, lnxg, zb);
    lstm(1, zb, NHID);
    lstm(2, h4 + (size_t)1 * S, NHID);
    lstm(3, h4 + (size_t)2 * S, NHID);
    conv(h4 + (size_t)3 * S, NHID, 1, pWout, CC, outc);
    output_k<<<CC * NN / 256, 256, 0, stream>>>(outc, boutW, images, xgen,
                                                gen, loss, t);
  }
}
